// Generator_40355512713722
// MI455X (gfx1250) — compile-verified, matrix operand order checked by reading, not run on hardware
//
#include <hip/hip_runtime.h>
#include <hip/hip_bf16.h>
#include <math.h>

#define IDIM  128
#define HD    256
#define BATCH 512
#define TT    128
#define GDIM  1024            // 4*HD
#define NROWS (BATCH * TT)    // 65536
#define OUTD  128

#define BM 128
#define BN 64
#define BK 32

typedef __attribute__((ext_vector_type(16))) __bf16 v16bf;
typedef __attribute__((ext_vector_type(8)))  __bf16 bf16x8;
typedef __attribute__((ext_vector_type(8)))  float  v8f;

__device__ __forceinline__ unsigned short f2bf(float f) {
  union { float f; unsigned u; } v; v.f = f;
  unsigned u = v.u;
  unsigned r = (u + 0x7FFFu + ((u >> 16) & 1u)) >> 16;  // RNE
  return (unsigned short)r;
}
__device__ __forceinline__ float bf2f(unsigned short b) {
  union { unsigned u; float f; } v; v.u = ((unsigned)b) << 16;
  return v.f;
}
__device__ __forceinline__ float sigm(float x) { return 1.0f / (1.0f + __expf(-x)); }

// ---------------------------------------------------------------------------
// Generic bf16 WMMA GEMM:  C[M,N] = A[M,K](bf16) * Bt[N,K]^T(bf16) + bias[N]
//                                   (+ optional bf16 addend Cin with row stride)
// mode 0: store f32 to outF
// mode 1: leaky_relu(0.1) then store bf16 to outB
// mode 2: store bf16 raw to outB
// Block tile 128x64 (8 waves, each wave 32x32 via 4 WMMA accumulators).
// Global->LDS staging via CDNA5 async DMA (global_load_async_to_lds_b128,
// ASYNCcnt-tracked). M multiple of 128, N multiple of 64, K multiple of 32.
// ---------------------------------------------------------------------------
__global__ __launch_bounds__(256)
void gemm_bf16_wmma(const unsigned short* __restrict__ A,
                    const unsigned short* __restrict__ Bt,
                    const float* __restrict__ bias,
                    const unsigned short* __restrict__ Cin,
                    long cinStride,
                    float* __restrict__ outF,
                    unsigned short* __restrict__ outB,
                    int M, int N, int K, int mode)
{
  // Single LDS object -> known byte offsets for the async DMA (base 0).
  __shared__ __align__(16) unsigned short Smem[(BM + BN) * BK];  // A then B
  unsigned short* As = Smem;            // LDS byte offset 0
  unsigned short* Bs = Smem + BM * BK;  // LDS byte offset 8192

  const int tid = threadIdx.x;

  // Visible LDS store the compiler cannot prove dead (mode is runtime, never
  // >= 256). Without any visible store the AMDGPU LDS optimizer folds all
  // loads from Smem to undef (the async-DMA writes live inside inline asm).
  if (mode >= 256) { Smem[tid] = (unsigned short)K; }

  const int lane  = tid & 31;
  const int wave  = tid >> 5;
  const int wm    = wave >> 1;       // 0..3 -> 32-row sub-tile
  const int wn    = wave & 1;        // 0..1 -> 32-col sub-tile
  const int laneR = lane & 15;
  const int kh    = (lane >> 4) * 8; // K-half select per WMMA bf16 layout

  const int blockN = blockIdx.x * BN;
  const int blockM = blockIdx.y * BM;

  // Loader mapping: 16B chunks; chunk c -> row c/4, col (c&3)*8.
  // A tile: 128x32 = 512 chunks -> thread loads chunks tid and tid+256.
  // B tile:  64x32 = 256 chunks -> thread loads chunk tid.
  const int ar0 = tid >> 2;
  const int ar1 = ar0 + 64;
  const int ac  = (tid & 3) * 8;

  const unsigned aoff0 = (unsigned)((ar0 * BK + ac) * 2);
  const unsigned aoff1 = (unsigned)((ar1 * BK + ac) * 2);
  const unsigned boff  = (unsigned)((BM * BK + ar0 * BK + ac) * 2);

  v8f acc[2][2] = {};

  for (int k0 = 0; k0 < K; k0 += BK) {
    const unsigned short* ga0 = &A [(long)(blockM + ar0) * K + k0 + ac];
    const unsigned short* ga1 = &A [(long)(blockM + ar1) * K + k0 + ac];
    const unsigned short* gb  = &Bt[(long)(blockN + ar0) * K + k0 + ac];
    asm volatile("global_load_async_to_lds_b128 %0, %1, off"
                 :: "v"(aoff0), "v"(ga0) : "memory");
    asm volatile("global_load_async_to_lds_b128 %0, %1, off"
                 :: "v"(aoff1), "v"(ga1) : "memory");
    asm volatile("global_load_async_to_lds_b128 %0, %1, off"
                 :: "v"(boff),  "v"(gb)  : "memory");
    asm volatile("s_wait_asynccnt 0" ::: "memory");
    __syncthreads();

    // A fragments: lane (l<16) row l, K = kh..kh+7 and kh+16..kh+23.
    // Volatile b128 LDS reads -> guaranteed ds_load_b128.
    v16bf afr[2], bfr[2];
    #pragma unroll
    for (int am = 0; am < 2; ++am) {
      const __bf16* ap = (const __bf16*)&As[(wm * 32 + am * 16 + laneR) * BK];
      bf16x8 lo = *(volatile const bf16x8*)(ap + kh);
      bf16x8 hi = *(volatile const bf16x8*)(ap + kh + 16);
      #pragma unroll
      for (int i = 0; i < 8; ++i) { afr[am][i] = lo[i]; afr[am][8 + i] = hi[i]; }
    }
    #pragma unroll
    for (int bn = 0; bn < 2; ++bn) {
      const __bf16* bp = (const __bf16*)&Bs[(wn * 32 + bn * 16 + laneR) * BK];
      bf16x8 lo = *(volatile const bf16x8*)(bp + kh);
      bf16x8 hi = *(volatile const bf16x8*)(bp + kh + 16);
      #pragma unroll
      for (int i = 0; i < 8; ++i) { bfr[bn][i] = lo[i]; bfr[bn][8 + i] = hi[i]; }
    }

    #pragma unroll
    for (int am = 0; am < 2; ++am)
      #pragma unroll
      for (int bn = 0; bn < 2; ++bn)
        acc[am][bn] = __builtin_amdgcn_wmma_f32_16x16x32_bf16(
            false, afr[am], false, bfr[bn], (short)0, acc[am][bn], false, false);

    __syncthreads();
  }

  // C/D layout: VGPR i, lanes 0-15 -> M=i, lanes 16-31 -> M=8+i; N=lane%16
  #pragma unroll
  for (int am = 0; am < 2; ++am) {
    const int mB = blockM + wm * 32 + am * 16 + (lane >> 4) * 8;
    #pragma unroll
    for (int bn = 0; bn < 2; ++bn) {
      const int n = blockN + wn * 32 + bn * 16 + laneR;
      #pragma unroll
      for (int i = 0; i < 8; ++i) {
        const long m = mB + i;
        float v = acc[am][bn][i] + bias[n];
        if (Cin) v += bf2f(Cin[m * cinStride + n]);
        if (mode == 0) {
          outF[m * (long)N + n] = v;
        } else {
          float w = v;
          if (mode == 1) w = (v > 0.0f) ? v : 0.1f * v;
          outB[m * (long)N + n] = f2bf(w);
        }
      }
    }
  }
}

// ---------------------------------------------------------------------------
// Weight transpose + f32->bf16:  W[K,N] -> Wt[N,K]
// ---------------------------------------------------------------------------
__global__ void transpose_bf16_kernel(const float* __restrict__ W,
                                      unsigned short* __restrict__ Wt,
                                      int K, int N)
{
  long idx = (long)blockIdx.x * blockDim.x + threadIdx.x;
  if (idx >= (long)K * N) return;
  int n = (int)(idx % N);
  int k = (int)(idx / N);
  Wt[(long)n * K + k] = f2bf(W[idx]);
}

// x0[B,T,ID] -> Xf bf16 (row b*T+t) and Xr bf16 (row b*T+t holds x0[b,T-1-t])
__global__ void convert_x_kernel(const float* __restrict__ x0,
                                 unsigned short* __restrict__ Xf,
                                 unsigned short* __restrict__ Xr)
{
  long idx = (long)blockIdx.x * blockDim.x + threadIdx.x;
  if (idx >= (long)NROWS * IDIM) return;
  int  col = (int)(idx & (IDIM - 1));
  long row = idx >> 7;               // IDIM = 128
  int  t   = (int)(row & (TT - 1));
  long b   = row >> 7;               // TT = 128
  Xf[idx] = f2bf(x0[idx]);
  long rrow = (b << 7) + (TT - 1 - t);
  Xr[idx] = f2bf(x0[(rrow << 7) + col]);
}

__global__ void zero_kernel(unsigned int* __restrict__ p, long n)
{
  long i = (long)blockIdx.x * blockDim.x + threadIdx.x;
  if (i < n) p[i] = 0u;
}

// ---------------------------------------------------------------------------
// LSTM cell elementwise, both directions. gates[2,512,1024] order f,i,a,o.
// c f32 [2,512,256]; h bf16 [2,512,256]; writes h into x1[b*T+t, dir*256+j].
// ---------------------------------------------------------------------------
__global__ void lstm_cell_kernel(const float* __restrict__ gates,
                                 float* __restrict__ c,
                                 unsigned short* __restrict__ h,
                                 unsigned short* __restrict__ x1,
                                 int t)
{
  int idx = blockIdx.x * blockDim.x + threadIdx.x;   // 2*512*256 = 262144
  if (idx >= 2 * BATCH * HD) return;
  int j   = idx & (HD - 1);
  int b   = (idx >> 8) & (BATCH - 1);
  int dir = idx >> 17;
  const float* g = gates + ((long)dir * BATCH + b) * GDIM;
  float f  = sigm(g[j]);
  float ii = sigm(g[HD + j]);
  float a  = tanhf(g[2 * HD + j]);
  float o  = sigm(g[3 * HD + j]);
  float cn = f * c[idx] + ii * a;
  c[idx] = cn;
  float hn = o * tanhf(cn);
  unsigned short hb = f2bf(hn);
  h[idx] = hb;
  x1[((long)b * TT + t) * (2 * HD) + (long)dir * HD + j] = hb;
}

// ---------------------------------------------------------------------------
// Head: sigmoid(0:64), softmax(64:72), softmax(72:88), softmax(88:128).
// One thread per (b,t) row.
// ---------------------------------------------------------------------------
__global__ void head_kernel(const float* __restrict__ x4, float* __restrict__ out)
{
  long r = (long)blockIdx.x * blockDim.x + threadIdx.x;
  if (r >= NROWS) return;
  const float* v = x4 + r * OUTD;
  float* o = out + r * OUTD;
  for (int i = 0; i < 64; ++i) o[i] = sigm(v[i]);
  const int segs[4] = {64, 72, 88, 128};
  for (int s = 0; s < 3; ++s) {
    int a = segs[s], b = segs[s + 1];
    float mx = -1e30f;
    for (int i = a; i < b; ++i) mx = fmaxf(mx, v[i]);
    float sum = 0.0f;
    for (int i = a; i < b; ++i) { float e = __expf(v[i] - mx); o[i] = e; sum += e; }
    float inv = 1.0f / sum;
    for (int i = a; i < b; ++i) o[i] *= inv;
  }
}

// ---------------------------------------------------------------------------
extern "C" void kernel_launch(void* const* d_in, const int* in_sizes, int n_in,
                              void* d_out, int out_size, void* d_ws, size_t ws_size,
                              hipStream_t stream)
{
  const float* x0   = (const float*)d_in[0];
  const float* Wi_f = (const float*)d_in[1];
  const float* bi_f = (const float*)d_in[2];
  const float* Wh_f = (const float*)d_in[3];
  const float* bh_f = (const float*)d_in[4];
  const float* Wi_r = (const float*)d_in[5];
  const float* bi_r = (const float*)d_in[6];
  const float* Wh_r = (const float*)d_in[7];
  const float* bh_r = (const float*)d_in[8];
  const float* W1   = (const float*)d_in[9];
  const float* b1   = (const float*)d_in[10];
  const float* W2   = (const float*)d_in[11];
  const float* b2   = (const float*)d_in[12];
  const float* W3   = (const float*)d_in[13];
  const float* b3   = (const float*)d_in[14];
  float* out = (float*)d_out;

  char* p = (char*)d_ws;
  auto carve = [&](size_t bytes) -> char* {
    char* r = p; p += (bytes + 255) & ~(size_t)255; return r;
  };

  unsigned short* Xf   = (unsigned short*)carve((size_t)NROWS * IDIM * 2);
  unsigned short* Xr   = (unsigned short*)carve((size_t)NROWS * IDIM * 2);
  unsigned short* WiTf = (unsigned short*)carve((size_t)GDIM * IDIM * 2);
  unsigned short* WiTr = (unsigned short*)carve((size_t)GDIM * IDIM * 2);
  unsigned short* WhTf = (unsigned short*)carve((size_t)GDIM * HD * 2);
  unsigned short* WhTr = (unsigned short*)carve((size_t)GDIM * HD * 2);
  unsigned short* W1T  = (unsigned short*)carve((size_t)HD * (2 * HD) * 2);
  unsigned short* W2T  = (unsigned short*)carve((size_t)HD * HD * 2);
  unsigned short* W3T  = (unsigned short*)carve((size_t)OUTD * HD * 2);
  unsigned short* Ginf = (unsigned short*)carve((size_t)NROWS * GDIM * 2);
  unsigned short* Ginr = (unsigned short*)carve((size_t)NROWS * GDIM * 2);
  float*          gate = (float*)carve((size_t)2 * BATCH * GDIM * 4);
  unsigned short* hbuf = (unsigned short*)carve((size_t)2 * BATCH * HD * 2);
  float*          cbuf = (float*)carve((size_t)2 * BATCH * HD * 4);
  unsigned short* x1   = (unsigned short*)carve((size_t)NROWS * (2 * HD) * 2);
  unsigned short* x2   = (unsigned short*)carve((size_t)NROWS * HD * 2);
  unsigned short* x3   = (unsigned short*)carve((size_t)NROWS * HD * 2);
  float*          x4   = (float*)carve((size_t)NROWS * OUTD * 4);

  const int THR = 256;
  auto blocks = [](long n, int t) { return (unsigned)((n + t - 1) / t); };

  // Weight transposes + bf16 conversion
  transpose_bf16_kernel<<<blocks((long)IDIM * GDIM, THR), THR, 0, stream>>>(Wi_f, WiTf, IDIM, GDIM);
  transpose_bf16_kernel<<<blocks((long)IDIM * GDIM, THR), THR, 0, stream>>>(Wi_r, WiTr, IDIM, GDIM);
  transpose_bf16_kernel<<<blocks((long)HD * GDIM, THR), THR, 0, stream>>>(Wh_f, WhTf, HD, GDIM);
  transpose_bf16_kernel<<<blocks((long)HD * GDIM, THR), THR, 0, stream>>>(Wh_r, WhTr, HD, GDIM);
  transpose_bf16_kernel<<<blocks((long)(2 * HD) * HD, THR), THR, 0, stream>>>(W1, W1T, 2 * HD, HD);
  transpose_bf16_kernel<<<blocks((long)HD * HD, THR), THR, 0, stream>>>(W2, W2T, HD, HD);
  transpose_bf16_kernel<<<blocks((long)HD * OUTD, THR), THR, 0, stream>>>(W3, W3T, HD, OUTD);

  // x0 -> bf16 (forward + time-reversed)
  convert_x_kernel<<<blocks((long)NROWS * IDIM, THR), THR, 0, stream>>>(x0, Xf, Xr);

  // zero h and c (contiguous carve region: h then c)
  long zeroWords = ((long)2 * BATCH * HD * 2 + (long)2 * BATCH * HD * 4) / 4;
  zero_kernel<<<blocks(zeroWords, THR), THR, 0, stream>>>((unsigned int*)hbuf, zeroWords);

  // Input projections: Gin = X @ Wi + bi  (bf16 out), M=65536, N=1024, K=128
  dim3 gIn(GDIM / BN, NROWS / BM);
  gemm_bf16_wmma<<<gIn, THR, 0, stream>>>(Xf, WiTf, bi_f, nullptr, 0, nullptr, Ginf,
                                          NROWS, GDIM, IDIM, 2);
  gemm_bf16_wmma<<<gIn, THR, 0, stream>>>(Xr, WiTr, bi_r, nullptr, 0, nullptr, Ginr,
                                          NROWS, GDIM, IDIM, 2);

  // Recurrent scan: 128 steps x (2 GEMM + 1 cell)
  dim3 gRec(GDIM / BN, BATCH / BM);
  const long ginStride = (long)TT * GDIM;  // row b -> element (b*T + t)*GDIM
  for (int t = 0; t < TT; ++t) {
    gemm_bf16_wmma<<<gRec, THR, 0, stream>>>(hbuf, WhTf, bh_f,
                                             Ginf + (long)t * GDIM, ginStride,
                                             gate, nullptr, BATCH, GDIM, HD, 0);
    gemm_bf16_wmma<<<gRec, THR, 0, stream>>>(hbuf + (long)BATCH * HD, WhTr, bh_r,
                                             Ginr + (long)t * GDIM, ginStride,
                                             gate + (long)BATCH * GDIM, nullptr,
                                             BATCH, GDIM, HD, 0);
    lstm_cell_kernel<<<blocks((long)2 * BATCH * HD, THR), THR, 0, stream>>>(gate, cbuf, hbuf, x1, t);
  }

  // MLP
  dim3 g2(HD / BN, NROWS / BM);
  gemm_bf16_wmma<<<g2, THR, 0, stream>>>(x1, W1T, b1, nullptr, 0, nullptr, x2,
                                         NROWS, HD, 2 * HD, 1);
  gemm_bf16_wmma<<<g2, THR, 0, stream>>>(x2, W2T, b2, nullptr, 0, nullptr, x3,
                                         NROWS, HD, HD, 1);
  dim3 g4(OUTD / BN, NROWS / BM);
  gemm_bf16_wmma<<<g4, THR, 0, stream>>>(x3, W3T, b3, nullptr, 0, x4, nullptr,
                                         NROWS, OUTD, HD, 0);

  head_kernel<<<blocks(NROWS, THR), THR, 0, stream>>>(x4, out);
}
